// CTRNN_50087908606434
// MI455X (gfx1250) — matrix operand order repaired and checked by reading, not compiled, not run on hardware
//
#include <hip/hip_runtime.h>
#include <hip/hip_bf16.h>
#include <stdint.h>

#define DT_C     0.02f
#define HIDDEN   512
#define TSTEPS   1024
#define NHALF    256      // N columns per workgroup (N split across 2 WGs)
#define MROWS    16       // batch rows per workgroup pair
#define JSTRIDE  520      // bf16 elems per J row in LDS (8-elem pad -> no bank conflict)
#define ASTRIDE  520      // bf16 elems per A row in LDS
#define NTHREADS 512      // 16 waves (wave32)
#define VCHUNK   32       // vel timesteps staged per LDS refill

typedef __attribute__((ext_vector_type(16))) __bf16        v16bf;
typedef __attribute__((ext_vector_type(8)))  float         v8f;
typedef __attribute__((ext_vector_type(8)))  unsigned int  v8u;
typedef __attribute__((ext_vector_type(4)))  unsigned int  u32x4;

__device__ __forceinline__ unsigned short f2bf(float f) {
  union { float f; unsigned int u; } v; v.f = f;
  unsigned int u = v.u;
  u += 0x7FFFu + ((u >> 16) & 1u);            // round-to-nearest-even
  return (unsigned short)(u >> 16);
}
__device__ __forceinline__ float bf2f(unsigned short u) {
  union { float f; unsigned int u; } v; v.u = ((unsigned int)u) << 16;
  return v.f;
}
__device__ __forceinline__ unsigned int pack2(float a, float b) {
  return (unsigned int)f2bf(a) | ((unsigned int)f2bf(b) << 16);
}

// Assemble a 16x(bf16) WMMA fragment from two 16B LDS loads.
__device__ __forceinline__ v16bf load_frag(const char* p0, const char* p1) {
  const uint4 a = *(const uint4*)p0;
  const uint4 b = *(const uint4*)p1;
  v8u r;
  r[0] = a.x; r[1] = a.y; r[2] = a.z; r[3] = a.w;
  r[4] = b.x; r[5] = b.y; r[6] = b.z; r[7] = b.w;
  return __builtin_bit_cast(v16bf, r);
}

__global__ void __launch_bounds__(512)
ctrnn_init_counters(int* cnt) {
  cnt[threadIdx.x] = 0;                       // 512 ints = 2048 B counter region
}

__global__ void __launch_bounds__(NTHREADS, 1)
ctrnn_main(const float* __restrict__ vel,     // [128, 1024]
           const float* __restrict__ Jg,      // [512, 512] row-major (n, k)
           const float* __restrict__ Bmat,    // [512]
           const float* __restrict__ Wro,     // [512]
           float* __restrict__ out,           // [128, 1024]
           int* __restrict__ cnt,
           unsigned short* __restrict__ xbuf) // [16 wg][2 bufs][4096] bf16 (frag layout)
{
  extern __shared__ char smem[];
  unsigned short* Jlds = (unsigned short*)smem;                          // 256*520 bf16
  unsigned short* Abuf = (unsigned short*)(smem + NHALF * JSTRIDE * 2);  // 16*520 bf16
  float* WroL = (float*)(smem + NHALF * JSTRIDE * 2 + MROWS * ASTRIDE * 2);
  float* vbuf = WroL + HIDDEN;                                           // 16*32 floats

  const int tid  = threadIdx.x;
  const int lane = tid & 31;
  const int w    = tid >> 5;                  // wave id == n-tile id (0..15)
  const int bg   = blockIdx.x >> 1;           // batch group 0..7
  const int nh   = blockIdx.x & 1;            // which N half
  const int n0g  = nh * NHALF;
  const int b0   = bg * MROWS;

  // ---- preamble: J half -> LDS as bf16 (n-major, padded stride) ----
  for (int i = tid; i < (NHALF * HIDDEN) / 4; i += NTHREADS) {
    const int e = i << 2;
    const int ln = e >> 9;                    // local n row
    const int k  = e & (HIDDEN - 1);
    const float4 f = *(const float4*)(Jg + (size_t)(n0g + ln) * HIDDEN + k);
    uint2 p; p.x = pack2(f.x, f.y); p.y = pack2(f.z, f.w);
    *(uint2*)((char*)Jlds + ln * (JSTRIDE * 2) + k * 2) = p;
  }
  // zero A buffer (tanh(h0) == 0)
  for (int i = tid; i < (MROWS * ASTRIDE) / 2; i += NTHREADS)
    ((unsigned int*)Abuf)[i] = 0u;
  if (tid < HIDDEN) WroL[tid] = Wro[tid];
  const float bm = Bmat[n0g + w * 16 + (lane & 15)];   // per-lane B column weight

  int* cnt_me  = cnt + (blockIdx.x) * 32;              // own cacheline each
  int* cnt_sib = cnt + (blockIdx.x ^ 1) * 32;
  __syncthreads();

  v8f h;
  #pragma unroll
  for (int r = 0; r < 8; ++r) h[r] = 0.f;

  const int rowsel = (lane < 16) ? 0 : 8;
  // A fragment base: row = lane&15; lanes>=16 take the second 16B of each 64B k-block
  const char* Abase = (const char*)Abuf + (lane & 15) * (ASTRIDE * 2) + ((lane < 16) ? 0 : 16);
  // B fragment base: column n = lane&15 of this wave's tile; lanes>=16 take the second 16B of each 64B k-block
  const char* Bbase = (const char*)Jlds + (w * 16 + (lane & 15)) * (JSTRIDE * 2) + ((lane < 16) ? 0 : 16);

  // exchange-chunk bases (fragment layout; 16B per lane, fully coalesced)
  unsigned short* myx0 = xbuf + (size_t)(blockIdx.x * 2 + 0) * (MROWS * NHALF);
  unsigned short* myx1 = xbuf + (size_t)(blockIdx.x * 2 + 1) * (MROWS * NHALF);
  const unsigned short* sx0 = xbuf + (size_t)((blockIdx.x ^ 1) * 2 + 0) * (MROWS * NHALF);
  const unsigned short* sx1 = xbuf + (size_t)((blockIdx.x ^ 1) * 2 + 1) * (MROWS * NHALF);
  const int fragoff = (w * 32 + lane) * 8;    // element offset of this lane's 8 values
  const int obase   = (1 - nh) * NHALF;       // sibling's column range in A buffer

  // vel staging indices: thread -> (row, t-within-chunk), coalesced per row
  const int vrow = tid >> 5;
  const int vtt  = tid & 31;

  #pragma unroll 1
  for (int t = 0; t < TSTEPS; ++t) {
    if ((t & (VCHUNK - 1)) == 0) {
      const float vv = vel[(size_t)(b0 + vrow) * TSTEPS + t + vtt];
      vbuf[vrow * VCHUNK + vtt] = vv;
    }

    // ---- GEMM: C(16x16) = tanh(h_{t-1})[16x512] x J_half^T tile [512x16] ----
    // Software-pipelined: fragments for kb+1 are in flight while WMMA kb executes.
    v8f c;
    #pragma unroll
    for (int r = 0; r < 8; ++r) c[r] = 0.f;
    v16bf af0 = load_frag(Abase, Abase + 32);
    v16bf bf0 = load_frag(Bbase, Bbase + 32);
    #pragma unroll
    for (int kb = 0; kb < HIDDEN / 32 - 1; ++kb) {
      const v16bf af1 = load_frag(Abase + (kb + 1) * 64, Abase + (kb + 1) * 64 + 32);
      const v16bf bf1 = load_frag(Bbase + (kb + 1) * 64, Bbase + (kb + 1) * 64 + 32);
      c = __builtin_amdgcn_wmma_f32_16x16x32_bf16(false, af0, false, bf0,
                                                  (short)0, c, false, false);
      af0 = af1; bf0 = bf1;
    }
    c = __builtin_amdgcn_wmma_f32_16x16x32_bf16(false, af0, false, bf0,
                                                (short)0, c, false, false);
    __syncthreads();   // all waves done reading A before it is overwritten

    // ---- leaky-integrator update, tanh, publish own half ----
    unsigned short th[8];
    #pragma unroll
    for (int r = 0; r < 8; ++r) {
      const int row = r + rowsel;
      const float v = vbuf[row * VCHUNK + (t & (VCHUNK - 1))];
      h[r] = h[r] * (1.0f - DT_C) + DT_C * (c[r] + v * bm);
      th[r] = f2bf(tanhf(h[r]));
      Abuf[row * ASTRIDE + n0g + w * 16 + (lane & 15)] = th[r];   // LDS (own half)
    }
    {   // one coalesced 16B store per lane, fragment layout
      u32x4 p;
      p[0] = (unsigned int)th[0] | ((unsigned int)th[1] << 16);
      p[1] = (unsigned int)th[2] | ((unsigned int)th[3] << 16);
      p[2] = (unsigned int)th[4] | ((unsigned int)th[5] << 16);
      p[3] = (unsigned int)th[6] | ((unsigned int)th[7] << 16);
      unsigned short* myx = (t & 1) ? myx1 : myx0;
      *(u32x4*)(myx + fragoff) = p;
    }
    __threadfence();
    __syncthreads();
    if (tid == 0) {
      __hip_atomic_fetch_add(cnt_me, 1, __ATOMIC_RELEASE, __HIP_MEMORY_SCOPE_AGENT);
      while (__hip_atomic_load(cnt_sib, __ATOMIC_ACQUIRE, __HIP_MEMORY_SCOPE_AGENT) < t + 1)
        __builtin_amdgcn_s_sleep(2);
    }
    __syncthreads();

    // ---- pull sibling half: 1 b128 load + 8 b16 LDS scatters per thread ----
    {
      const unsigned short* sx = (t & 1) ? sx1 : sx0;
      const u32x4 d = *(const u32x4*)(sx + tid * 8);
      const int l2   = tid & 31;
      const int w2   = tid >> 5;
      const int colL = obase + w2 * 16 + (l2 & 15);
      const int rb   = (l2 < 16) ? 0 : 8;
      #pragma unroll
      for (int j = 0; j < 4; ++j) {
        Abuf[(rb + 2 * j)     * ASTRIDE + colL] = (unsigned short)(d[j] & 0xFFFFu);
        Abuf[(rb + 2 * j + 1) * ASTRIDE + colL] = (unsigned short)(d[j] >> 16);
      }
    }
    __syncthreads();

    // ---- readout: wave w computes out[b0+w, t] (only the nh==0 sibling) ----
    if (nh == 0 && w < MROWS) {
      float acc = 0.f;
      const char* ap = (const char*)Abuf + w * (ASTRIDE * 2) + lane * 32;
      const uint4 a0 = *(const uint4*)ap;
      const uint4 a1 = *(const uint4*)(ap + 16);
      const float* wp = WroL + lane * 16;
      const float4 w0 = *(const float4*)(wp);
      const float4 w1 = *(const float4*)(wp + 4);
      const float4 w2 = *(const float4*)(wp + 8);
      const float4 w3 = *(const float4*)(wp + 12);
      acc += bf2f((unsigned short)(a0.x & 0xFFFFu)) * w0.x + bf2f((unsigned short)(a0.x >> 16)) * w0.y;
      acc += bf2f((unsigned short)(a0.y & 0xFFFFu)) * w0.z + bf2f((unsigned short)(a0.y >> 16)) * w0.w;
      acc += bf2f((unsigned short)(a0.z & 0xFFFFu)) * w1.x + bf2f((unsigned short)(a0.z >> 16)) * w1.y;
      acc += bf2f((unsigned short)(a0.w & 0xFFFFu)) * w1.z + bf2f((unsigned short)(a0.w >> 16)) * w1.w;
      acc += bf2f((unsigned short)(a1.x & 0xFFFFu)) * w2.x + bf2f((unsigned short)(a1.x >> 16)) * w2.y;
      acc += bf2f((unsigned short)(a1.y & 0xFFFFu)) * w2.z + bf2f((unsigned short)(a1.y >> 16)) * w2.w;
      acc += bf2f((unsigned short)(a1.z & 0xFFFFu)) * w3.x + bf2f((unsigned short)(a1.z >> 16)) * w3.y;
      acc += bf2f((unsigned short)(a1.w & 0xFFFFu)) * w3.z + bf2f((unsigned short)(a1.w >> 16)) * w3.w;
      #pragma unroll
      for (int off = 16; off > 0; off >>= 1)
        acc += __shfl_down(acc, off, 32);
      if (lane == 0) out[(size_t)(b0 + w) * TSTEPS + t] = acc;
    }
    // No extra barrier: next-iteration A writes are fenced by the post-GEMM barrier.
  }
}

extern "C" void kernel_launch(void* const* d_in, const int* in_sizes, int n_in,
                              void* d_out, int out_size, void* d_ws, size_t ws_size,
                              hipStream_t stream) {
  const float* vel  = (const float*)d_in[0];   // [128,1024,1]
  const float* J    = (const float*)d_in[1];   // [512,512]
  const float* Bm   = (const float*)d_in[2];   // [512,1]
  const float* Wro  = (const float*)d_in[3];   // [1,512]
  float* out        = (float*)d_out;           // [128,1024,1]

  int* cnt = (int*)d_ws;                                        // 2048 B of counters
  unsigned short* xbuf = (unsigned short*)((char*)d_ws + 4096); // 256 KB exchange

  ctrnn_init_counters<<<1, 512, 0, stream>>>(cnt);

  const size_t shmem = (size_t)(NHALF * JSTRIDE * 2)   // J half, bf16: 266240 B
                     + (size_t)(MROWS * ASTRIDE * 2)   // A buffer:       16640 B
                     + (size_t)(HIDDEN * 4)            // W_ro:            2048 B
                     + (size_t)(MROWS * VCHUNK * 4);   // vel chunk:       2048 B
  ctrnn_main<<<16, NTHREADS, shmem, stream>>>(vel, J, Bm, Wro, out, cnt, xbuf);
}
